// SinkhornPermutation_15247133901234
// MI455X (gfx1250) — matrix operand order repaired and checked by reading, not compiled, not run on hardware
//
#include <hip/hip_runtime.h>
#include <hip/hip_bf16.h>
#include <math.h>

// Sizes from the reference
#define S_ 4
#define B_ 8
#define D_ 1024
#define N_ 1024
#define NMAT (S_*B_)                 // 32 matrices
#define NN   ((size_t)N_*N_)         // 1M elements per matrix
#define SBNN ((size_t)NMAT*NN)       // 32M elements
#define EPSV 1e-10f
// TEMP == 1.0 -> the division is a no-op and is elided.

typedef __attribute__((ext_vector_type(16))) __bf16 v16bf;
typedef __attribute__((ext_vector_type(8)))  __bf16 bf16x8;
typedef __attribute__((ext_vector_type(8)))  float  v8f;
typedef __attribute__((ext_vector_type(4)))  int    v4i;

// AS-qualified v4i, per the builtin's signature (param0: v4i __device__*,
// param1: v4i __shared__*)
typedef __attribute__((address_space(1))) v4i av4i_g;
typedef __attribute__((address_space(3))) v4i av4i_l;

// Use CDNA5 async global->LDS DMA (ASYNCcnt) when the toolchain exposes it.
#if __has_builtin(__builtin_amdgcn_global_load_async_to_lds_b128) && \
    __has_builtin(__builtin_amdgcn_s_wait_asynccnt)
#define SK_ASYNC 1
#else
#define SK_ASYNC 0
#endif

#if SK_ASYNC
__device__ __forceinline__ void async_b128(const void* g, void* l) {
  __builtin_amdgcn_global_load_async_to_lds_b128(
      (av4i_g*)(v4i*)g, (av4i_l*)(v4i*)l, 0, 0);
}
#endif

__device__ __forceinline__ unsigned short f2bf(float f) {
  // float -> bf16 round-to-nearest-even via bit trick
  unsigned u = __float_as_uint(f);
  u += 0x7FFFu + ((u >> 16) & 1u);
  return (unsigned short)(u >> 16);
}

// ---------------------------------------------------------------------------
// 1) la[s,b,i,j] = log_alpha[i,j] + gumbel(u[s,b,i,j])
// ---------------------------------------------------------------------------
__global__ void sk_gumbel_init(const float* __restrict__ log_alpha,
                               const float* __restrict__ u,
                               float* __restrict__ la) {
  size_t i = (size_t)blockIdx.x * blockDim.x + threadIdx.x;
  float uu = u[i] + EPSV;
  float g  = -__logf(EPSV - __logf(uu));
  la[i] = log_alpha[i & (NN - 1)] + g;   // broadcast log_alpha over (S,B)
}

// ---------------------------------------------------------------------------
// 2a) column normalization: la -= logsumexp over rows (axis -2), per column.
// ---------------------------------------------------------------------------
__global__ void sk_col_norm(float* __restrict__ la) {
  const int z  = blockIdx.y;
  const int cc = threadIdx.x & 63;
  const int cr = threadIdx.x >> 6;              // 0..3
  const int c  = blockIdx.x * 64 + cc;
  float* mat = la + (size_t)z * NN;

  float m = -INFINITY, s = 0.f;                 // online logsumexp
  for (int r = cr; r < N_; r += 4) {
    float x = mat[(size_t)r * N_ + c];
    if (x > m) { s = s * __expf(m - x) + 1.f; m = x; }
    else       { s += __expf(x - m); }
  }
  __shared__ float sm[4][64], ss[4][64], slse[64];
  sm[cr][cc] = m; ss[cr][cc] = s;
  __syncthreads();
  if (cr == 0) {
    float M = sm[0][cc], Ssum = ss[0][cc];
    #pragma unroll
    for (int k = 1; k < 4; ++k) {
      float m2 = sm[k][cc], s2 = ss[k][cc];
      float mn = fmaxf(M, m2);
      Ssum = Ssum * __expf(M - mn) + s2 * __expf(m2 - mn);
      M = mn;
    }
    slse[cc] = M + __logf(Ssum);
  }
  __syncthreads();
  float lse = slse[cc];
  for (int r = cr; r < N_; r += 4)
    mat[(size_t)r * N_ + c] -= lse;
}

// ---------------------------------------------------------------------------
// 2b) row normalization (one wave32 per row). Final iteration fuses
// exp() + bf16 conversion, writing P (row-major) with coalesced stores.
// ---------------------------------------------------------------------------
__global__ void sk_row_norm(float* __restrict__ la,
                            unsigned short* __restrict__ P, int last) {
  const int z    = blockIdx.y;
  const int w    = threadIdx.x >> 5;
  const int lane = threadIdx.x & 31;
  const int row  = blockIdx.x * 8 + w;
  float* r = la + (size_t)z * NN + (size_t)row * N_;

  float m = -INFINITY, s = 0.f;
  for (int j = lane; j < N_; j += 32) {
    float x = r[j];
    if (x > m) { s = s * __expf(m - x) + 1.f; m = x; }
    else       { s += __expf(x - m); }
  }
  #pragma unroll
  for (int off = 16; off > 0; off >>= 1) {       // wave32 butterfly
    float m2 = __shfl_xor(m, off, 32);
    float s2 = __shfl_xor(s, off, 32);
    float mn = fmaxf(m, m2);
    s = s * __expf(m - mn) + s2 * __expf(m2 - mn);
    m = mn;
  }
  float lse = m + __logf(s);
  if (last) {
    unsigned short* pr = P + (size_t)z * NN + (size_t)row * N_;
    for (int j = lane; j < N_; j += 32)
      pr[j] = f2bf(__expf(r[j] - lse));
  } else {
    for (int j = lane; j < N_; j += 32)
      r[j] -= lse;
  }
}

// ---------------------------------------------------------------------------
// 2c) one-shot tiled transpose: Pt[n][k] = P[k][n]. Coalesced both sides via
// a 64x64 LDS tile (row stride 72 keeps 16B alignment for b128 LDS loads).
// ---------------------------------------------------------------------------
__global__ void sk_transpose_p(const unsigned short* __restrict__ P,
                               unsigned short* __restrict__ Pt) {
  __shared__ __align__(64) unsigned short t[64][72];
  const int z  = blockIdx.z;
  const int r0 = blockIdx.y * 64;    // source rows (k)
  const int c0 = blockIdx.x * 64;    // source cols (n)
  const unsigned short* src = P  + (size_t)z * NN;
  unsigned short*       dst = Pt + (size_t)z * NN;

  const int r = threadIdx.x >> 2;            // 0..63
  const int c = (threadIdx.x & 3) * 16;      // 0,16,32,48
  uint4 lo = *(const uint4*)&src[(size_t)(r0 + r) * N_ + c0 + c];
  uint4 hi = *(const uint4*)&src[(size_t)(r0 + r) * N_ + c0 + c + 8];
  unsigned v32[8] = {lo.x, lo.y, lo.z, lo.w, hi.x, hi.y, hi.z, hi.w};
  #pragma unroll
  for (int j = 0; j < 8; ++j) {
    t[c + 2*j    ][r] = (unsigned short)(v32[j] & 0xFFFFu);
    t[c + 2*j + 1][r] = (unsigned short)(v32[j] >> 16);
  }
  __syncthreads();
  // write rows of Pt: row n = c0 + r, cols k = r0 + c
  *(bf16x8*)&dst[(size_t)(c0 + r) * N_ + r0 + c]     = *(const bf16x8*)&t[r][c];
  *(bf16x8*)&dst[(size_t)(c0 + r) * N_ + r0 + c + 8] = *(const bf16x8*)&t[r][c + 8];
}

// ---------------------------------------------------------------------------
// 3) x fp32 -> bf16
// ---------------------------------------------------------------------------
__global__ void sk_conv_x(const float* __restrict__ x, unsigned short* __restrict__ xb) {
  size_t i = (size_t)blockIdx.x * blockDim.x + threadIdx.x;
  xb[i] = f2bf(x[i]);
}

// ---------------------------------------------------------------------------
// GEMM tile staging: both tiles are plain row-major b128 copies.
// Async path: 3 x GLOBAL_LOAD_ASYNC_TO_LDS_B128 per thread (ASYNCcnt).
// ---------------------------------------------------------------------------
__device__ __forceinline__ void stage_tiles(const unsigned short* __restrict__ srcA,
                                            const unsigned short* __restrict__ srcB,
                                            unsigned short (*la)[32],
                                            unsigned short (*lb)[32],
                                            int tid) {
  const int rowA = tid >> 2, colA = (tid & 3) * 8;   // 64 x 32 A tile
  const int rowB = tid >> 1, colB = (tid & 1) * 16;  // 128 x 32 B tile (Pt rows)
#if SK_ASYNC
  async_b128(srcA + (size_t)rowA * N_ + colA,     &la[rowA][colA]);
  async_b128(srcB + (size_t)rowB * N_ + colB,     &lb[rowB][colB]);
  async_b128(srcB + (size_t)rowB * N_ + colB + 8, &lb[rowB][colB + 8]);
#else
  *(bf16x8*)&la[rowA][colA]     = *(const bf16x8*)(srcA + (size_t)rowA * N_ + colA);
  *(bf16x8*)&lb[rowB][colB]     = *(const bf16x8*)(srcB + (size_t)rowB * N_ + colB);
  *(bf16x8*)&lb[rowB][colB + 8] = *(const bf16x8*)(srcB + (size_t)rowB * N_ + colB + 8);
#endif
}

// ---------------------------------------------------------------------------
// 4) out[z] = x[b] @ P[z]; bf16 WMMA, f32 accumulate, double-buffered LDS.
// Block tile 64(M) x 128(N), 8 waves in 2x4, wave tile 32x32 (4 WMMA accs).
// ---------------------------------------------------------------------------
__global__ void __launch_bounds__(256)
sk_gemm_bf16(const unsigned short* __restrict__ xb,
             const unsigned short* __restrict__ Pt,
             float* __restrict__ out) {
  __shared__ __align__(64) unsigned short lds_a[2][64][32];    // [buf][m][k]
  __shared__ __align__(64) unsigned short lds_b[2][128][32];   // [buf][n][k]

  const int z    = blockIdx.z;
  const int b    = z & (B_ - 1);
  const int m0   = blockIdx.y * 64;
  const int n0   = blockIdx.x * 128;
  const int tid  = threadIdx.x;
  const int wave = tid >> 5, lane = tid & 31;
  const int wm   = wave >> 2, wn = wave & 3;                // 2 x 4 wave grid
  const int l16  = lane & 15;
  const int koff  = (lane < 16) ? 0 : 8;                    // A frag K offset
  const int khalf = (lane < 16) ? 0 : 16;                   // unused; B frag also uses koff
  const unsigned short* Abase = xb + (size_t)b * D_ * N_ + (size_t)m0 * N_;
  const unsigned short* Bbase = Pt + (size_t)z * NN       + (size_t)n0 * N_;

  v8f acc[2][2] = {};

  stage_tiles(Abase, Bbase, lds_a[0], lds_b[0], tid);       // prologue: tile 0

  for (int k0 = 0; k0 < N_; k0 += 32) {
    const int cur = (k0 >> 5) & 1;
    if (k0 + 32 < N_)
      stage_tiles(Abase + k0 + 32, Bbase + k0 + 32,
                  lds_a[cur ^ 1], lds_b[cur ^ 1], tid);
#if SK_ASYNC
    if (k0 + 32 < N_) __builtin_amdgcn_s_wait_asynccnt(3);  // tile k complete
    else              __builtin_amdgcn_s_wait_asynccnt(0);
#endif
    __syncthreads();

    // fragments (ISA 7.12.2 16-bit layouts)
    v16bf afrag[2], bfrag[2];
    #pragma unroll
    for (int t = 0; t < 2; ++t) {
      int mr = wm * 32 + t * 16 + l16;
      bf16x8 aLo = *(const bf16x8*)&lds_a[cur][mr][koff];        // K koff..+7
      bf16x8 aHi = *(const bf16x8*)&lds_a[cur][mr][koff + 16];   // K koff+16..+23
      v16bf a;
      #pragma unroll
      for (int e = 0; e < 8; ++e) { a[e] = aLo[e]; a[e + 8] = aHi[e]; }
      afrag[t] = a;
      int nc = wn * 32 + t * 16 + l16;
      bf16x8 bLo = *(const bf16x8*)&lds_b[cur][nc][koff];        // K koff..+7
      bf16x8 bHi = *(const bf16x8*)&lds_b[cur][nc][koff + 16];   // K koff+16..+23
      v16bf bv;
      #pragma unroll
      for (int e = 0; e < 8; ++e) { bv[e] = bLo[e]; bv[e + 8] = bHi[e]; }
      bfrag[t] = bv;
    }
    #pragma unroll
    for (int tm = 0; tm < 2; ++tm)
      #pragma unroll
      for (int tn = 0; tn < 2; ++tn)
        acc[tm][tn] = __builtin_amdgcn_wmma_f32_16x16x32_bf16(
            false, afrag[tm], false, bfrag[tn], (short)0, acc[tm][tn],
            false, false);
    __syncthreads();
  }

  // store C: lane 0-15 rows r, lanes 16-31 rows r+8; col = lane%16
  const int rbase = (lane < 16) ? 0 : 8;
  #pragma unroll
  for (int tm = 0; tm < 2; ++tm) {
    int d0 = m0 + wm * 32 + tm * 16 + rbase;
    #pragma unroll
    for (int tn = 0; tn < 2; ++tn) {
      int nc = n0 + wn * 32 + tn * 16 + l16;
      #pragma unroll
      for (int rr = 0; rr < 8; ++rr)
        out[(size_t)z * D_ * N_ + (size_t)(d0 + rr) * N_ + nc] = acc[tm][tn][rr];
    }
  }
}

// ---------------------------------------------------------------------------
extern "C" void kernel_launch(void* const* d_in, const int* in_sizes, int n_in,
                              void* d_out, int out_size, void* d_ws, size_t ws_size,
                              hipStream_t stream) {
  const float* x   = (const float*)d_in[0];   // (B,D,N)
  const float* la0 = (const float*)d_in[1];   // (N,N)
  const float* u   = (const float*)d_in[2];   // (S,B,N,N)
  float* out = (float*)d_out;                 // (S,B,D,N)

  // Workspace: la fp32 (128MB) | P bf16 (64MB) | x bf16 (16MB).
  // Pt (bf16, 64MB) aliases la, which is dead after the last row_norm.
  float* la = (float*)d_ws;
  unsigned short* P  = (unsigned short*)((char*)d_ws + SBNN * sizeof(float));
  unsigned short* xb = (unsigned short*)((char*)P + SBNN * sizeof(unsigned short));
  unsigned short* Pt = (unsigned short*)d_ws;

  sk_gumbel_init<<<(unsigned)(SBNN / 256), 256, 0, stream>>>(la0, u, la);
  for (int it = 0; it < 20; ++it) {
    sk_col_norm<<<dim3(N_ / 64, NMAT), 256, 0, stream>>>(la);
    sk_row_norm<<<dim3(N_ / 8, NMAT), 256, 0, stream>>>(la, P, it == 19);
  }
  sk_transpose_p<<<dim3(N_ / 64, N_ / 64, NMAT), 256, 0, stream>>>(P, Pt);
  sk_conv_x<<<(unsigned)(((size_t)B_ * D_ * N_) / 256), 256, 0, stream>>>(x, xb);
  sk_gemm_bf16<<<dim3(N_ / 128, D_ / 64, NMAT), 256, 0, stream>>>(xb, Pt, out);
}
